// BidirectionAttention_65644280152675
// MI455X (gfx1250) — matrix operand order repaired and checked by reading, not compiled, not run on hardware
//
#include <hip/hip_runtime.h>
#include <math.h>

// ---------------------------------------------------------------------------
// BidirectionAttention for MI455X (gfx1250, wave32, WMMA 16x16x32 f16 + TDM)
// ---------------------------------------------------------------------------

typedef __attribute__((ext_vector_type(16))) _Float16 v16h;
typedef __attribute__((ext_vector_type(8)))  _Float16 v8h;
typedef __attribute__((ext_vector_type(8)))  float    v8f;
typedef __attribute__((ext_vector_type(4)))  int      i32x4;
typedef __attribute__((ext_vector_type(8)))  int      i32x8;

#define HEADS 4
#define DH    64
#define INNER 256      // HEADS * DH
#define BATCH 2
#define FEATC 128
#define MAPC  64
#define OUTC  128
#define DD    24
#define NN    13824    // 24^3
#define MM    512      // 8^3
#define NP    27       // N partitions for map-direction attention
#define PS    512      // partition size: NP*PS == NN

#define WMMA_F16(a, b, c) \
  __builtin_amdgcn_wmma_f32_16x16x32_f16(false, (a), false, (b), (short)0, (c), false, false)

// A-operand (16x32 f16, rows striped in-lane): lane holds row (row0+lo),
// element e -> K = k0 + (e<8?0:16) + hi*8 + (e&7). Two 16B contiguous loads.
__device__ __forceinline__ v16h load_a16(const _Float16* p, int ld, int row0,
                                         int k0, int lo, int hi) {
  const _Float16* r = p + (size_t)(row0 + lo) * ld + k0 + hi * 8;
  v8h x0 = *(const v8h*)(r);
  v8h x1 = *(const v8h*)(r + 16);
  v16h a;
#pragma unroll
  for (int i = 0; i < 8; ++i) { a[i] = x0[i]; a[i + 8] = x1[i]; }
  return a;
}

// B-operand (32x16 f16): source stored transposed [col][k] row-major;
// element e -> K = k0 + (e<8?0:16) + hi*8 + (e&7). Two 16B contiguous loads.
__device__ __forceinline__ v16h load_b16(const _Float16* p, int ld, int col0,
                                         int k0, int lo, int hi) {
  const _Float16* r = p + (size_t)(col0 + lo) * ld + k0 + hi * 8;
  v8h x0 = *(const v8h*)(r);
  v8h x1 = *(const v8h*)(r + 16);
  return __builtin_shufflevector(x0, x1, 0, 1, 2, 3, 4, 5, 6, 7, 8, 9, 10, 11, 12, 13, 14, 15);
}

// Two 16x16 score tiles (A rows r0.. and r0+16.., K = DH = 64), scaled.
__device__ __forceinline__ void score_pair(const _Float16* arows, int r0,
                                           v16h b0, v16h b1, int lo, int hi,
                                           v8f& t0, v8f& t1) {
  v8f z0 = {}, z1 = {};
  z0 = WMMA_F16(load_a16(arows, DH, r0, 0, lo, hi), b0, z0);
  z0 = WMMA_F16(load_a16(arows, DH, r0, 32, lo, hi), b1, z0);
  z1 = WMMA_F16(load_a16(arows, DH, r0 + 16, 0, lo, hi), b0, z1);
  z1 = WMMA_F16(load_a16(arows, DH, r0 + 16, 32, lo, hi), b1, z1);
#pragma unroll
  for (int r = 0; r < 8; ++r) { z0[r] *= 0.125f; z1[r] *= 0.125f; }
  t0 = z0; t1 = z1;
}

// TDM 2D tile load global->LDS, 2-byte elements, 2-SGPR-group descriptor
// (groups 2/3 omitted => up to-2D form). Issue from one wave only; complete
// with s_wait_tensorcnt before publishing via workgroup barrier.
__device__ __forceinline__ void tdm_load_2d(void* lds_dst, const void* gsrc,
                                            unsigned width_elems, unsigned rows,
                                            unsigned long long stride_elems) {
  unsigned long long ga = (unsigned long long)gsrc;
  unsigned lds = (unsigned)(unsigned long long)lds_dst;  // generic low 32b = LDS offset
  i32x4 g0;
  g0[0] = 1;                                              // count=1, user D#
  g0[1] = (int)lds;                                       // lds_addr (bytes)
  g0[2] = (int)(unsigned)(ga & 0xFFFFFFFFull);            // global_addr[31:0]
  g0[3] = (int)((unsigned)((ga >> 32) & 0x1FFFFFFull) | 0x80000000u); // addr[56:32]|type=2
  i32x8 g1;
  g1[0] = (int)(1u << 16);                                // data_size=1 (2 bytes)
  g1[1] = (int)((width_elems & 0xFFFFu) << 16);           // tensor_dim0[15:0]
  g1[2] = (int)((width_elems >> 16) | ((rows & 0xFFFFu) << 16));       // dim0 hi | dim1 lo
  g1[3] = (int)((rows >> 16) | ((width_elems & 0xFFFFu) << 16));       // dim1 hi | tile_dim0
  g1[4] = (int)(rows & 0xFFFFu);                          // tile_dim1 (tile_dim2=0)
  g1[5] = (int)(unsigned)(stride_elems & 0xFFFFFFFFull);  // dim0_stride[31:0]
  g1[6] = (int)(unsigned)((stride_elems >> 32) & 0xFFFFull); // dim0_stride[47:32]
  g1[7] = 0;                                              // dim1_stride (unused, 2D tile)
  asm volatile("tensor_load_to_lds %0, %1" :: "s"(g0), "s"(g1) : "memory");
}

__device__ __forceinline__ void tdm_wait() {
  asm volatile("s_wait_tensorcnt 0x0" ::: "memory");
}

// ---------------------------------------------------------------------------
// K0: convert pointwise weight matrices to f16 (row-major [o][c])
__global__ void k_cvt_weights(const float* __restrict__ pwqv,
                              const float* __restrict__ pwout,
                              _Float16* __restrict__ pwqv_h,
                              _Float16* __restrict__ pwout_h) {
  int t = blockIdx.x * blockDim.x + threadIdx.x;
  if (t < 2 * INNER * FEATC) pwqv_h[t] = (_Float16)pwqv[t];
  if (t < OUTC * INNER)      pwout_h[t] = (_Float16)pwout[t];
}

// K1/K7: depthwise 3x3x3 SAME conv; x: [B,C,NN] f32, w: [C,27],
// out f16 [b][n][C] (c contiguous -> WMMA B-operand friendly)
__global__ void k_dw3(const float* __restrict__ x, const float* __restrict__ w,
                      _Float16* __restrict__ out, int C) {
  int t = blockIdx.x * blockDim.x + threadIdx.x;
  if (t >= BATCH * C * NN) return;
  int n = t % NN; int bc = t / NN; int c = bc % C; int b = bc / C;
  int xx0 = n % DD, yy0 = (n / DD) % DD, zz0 = n / (DD * DD);
  const float* src = x + (size_t)(b * C + c) * NN;
  const float* wc = w + c * 27;
  float acc = 0.f;
  for (int dz = -1; dz <= 1; ++dz) {
    int z = zz0 + dz; if (z < 0 || z >= DD) continue;
    for (int dy = -1; dy <= 1; ++dy) {
      int y = yy0 + dy; if (y < 0 || y >= DD) continue;
      for (int dx = -1; dx <= 1; ++dx) {
        int xc = xx0 + dx; if (xc < 0 || xc >= DD) continue;
        acc += wc[(dz + 1) * 9 + (dy + 1) * 3 + (dx + 1)] *
               src[(z * DD + y) * DD + xc];
      }
    }
  }
  out[((size_t)b * NN + n) * C + c] = (_Float16)acc;
}

// K2: qv pointwise GEMM via WMMA: [512 x 128] @ [128 x NN]; writes
// fq [b,h,n,d] and fvT [b,h,d,n] (f16), channel c -> (d=c/4, h=c%4)
__global__ void k_qv_gemm(const _Float16* __restrict__ pwqv_h,
                          const _Float16* __restrict__ dwfeat,
                          _Float16* __restrict__ fq,
                          _Float16* __restrict__ fvT) {
  int blk = blockIdx.x;
  int nt = blk % (NN / 16); blk /= (NN / 16);
  int ot = blk % 32; int b = blk / 32;
  int lane = threadIdx.x & 31, lo = lane & 15, hi = lane >> 4;
  const _Float16* bsrc = dwfeat + (size_t)b * NN * FEATC;
  v8f acc = {};
#pragma unroll
  for (int kc = 0; kc < FEATC; kc += 32) {
    v16h a  = load_a16(pwqv_h, FEATC, ot * 16, kc, lo, hi);
    v16h bm = load_b16(bsrc, FEATC, nt * 16, kc, lo, hi);
    acc = WMMA_F16(a, bm, acc);
  }
  int n = nt * 16 + lo;
#pragma unroll
  for (int r = 0; r < 8; ++r) {
    int o = ot * 16 + hi * 8 + r;
    _Float16 v = (_Float16)acc[r];
    if (o < INNER) {
      int d = o >> 2, h = o & 3;
      fq[(((size_t)b * HEADS + h) * NN + n) * DH + d] = v;
    } else {
      int ov = o - INNER; int d = ov >> 2, h = ov & 3;
      fvT[(((size_t)b * HEADS + h) * DH + d) * NN + n] = v;
    }
  }
}

// K3: map qv projection (tiny)
__global__ void k_map_qv(const float* __restrict__ w,
                         const float* __restrict__ smap,
                         _Float16* __restrict__ mq,
                         _Float16* __restrict__ mvT) {
  int t = blockIdx.x * blockDim.x + threadIdx.x;
  if (t >= BATCH * 2 * INNER * MM) return;
  int m = t % MM; int bo = t / MM; int o = bo % (2 * INNER); int b = bo / (2 * INNER);
  const float* s = smap + (size_t)b * MAPC * MM;
  const float* wr = w + (size_t)o * MAPC;
  float acc = 0.f;
  for (int c = 0; c < MAPC; ++c) acc += wr[c] * s[c * MM + m];
  _Float16 v = (_Float16)acc;
  if (o < INNER) {
    int d = o >> 2, h = o & 3;
    mq[(((size_t)b * HEADS + h) * MM + m) * DH + d] = v;
  } else {
    int ov = o - INNER; int d = ov >> 2, h = ov & 3;
    mvT[(((size_t)b * HEADS + h) * DH + d) * MM + m] = v;
  }
}

// K5: feat direction. 8 waves/block, each wave one 16-row n-tile. mq + mvT
// (64KB+64KB) staged once per block into LDS via TDM; two-sweep flash softmax
// over m (row index = lane&15), then P@mv with the transposed-tile layout
// trick (score C-layout == next WMMA's A-layout).
__global__ void k_feat_attn(const _Float16* __restrict__ fq,
                            const _Float16* __restrict__ mq,
                            const _Float16* __restrict__ mvT,
                            float* __restrict__ featattn) {
  __shared__ __align__(32) _Float16 s_mq[MM * DH];   // 64KB
  __shared__ __align__(32) _Float16 s_mvT[DH * MM];  // 64KB
  int blk = blockIdx.x;                  // bh*(NN/128) + g
  int g8 = blk % (NN / 128); int bh = blk / (NN / 128);
  int h = bh % HEADS, b = bh / HEADS;
  int wid = threadIdx.x >> 5;
  int nt = g8 * 8 + wid;
  int lane = threadIdx.x & 31, lo = lane & 15, hi = lane >> 4;
  const _Float16* fq_bh  = fq  + (size_t)bh * NN * DH;
  const _Float16* mq_bh  = mq  + (size_t)bh * MM * DH;
  const _Float16* mvT_bh = mvT + (size_t)bh * DH * MM;

  if (wid == 0) {
    tdm_load_2d(s_mq, mq_bh, DH, MM, DH);     // [m][d], contiguous
    tdm_load_2d(s_mvT, mvT_bh, MM, DH, MM);   // [d][m], contiguous
    tdm_wait();
  }
  __syncthreads();

  v16h bq0 = load_b16(fq_bh, DH, nt * 16, 0, lo, hi);
  v16h bq1 = load_b16(fq_bh, DH, nt * 16, 32, lo, hi);

  float mx = -1e30f, sm = 0.f;
  for (int mc = 0; mc < MM; mc += 32) {
    v8f t0, t1;
    score_pair(s_mq, mc, bq0, bq1, lo, hi, t0, t1);
    float cmax = mx;
#pragma unroll
    for (int r = 0; r < 8; ++r) cmax = fmaxf(cmax, fmaxf(t0[r], t1[r]));
    cmax = fmaxf(cmax, __shfl_xor(cmax, 16));
    float cs = 0.f;
#pragma unroll
    for (int r = 0; r < 8; ++r) cs += __expf(t0[r] - cmax) + __expf(t1[r] - cmax);
    cs += __shfl_xor(cs, 16);
    sm = sm * __expf(mx - cmax) + cs;
    mx = cmax;
  }
  float inv = 1.f / sm;

  v8f zero = {};
  v8f acc[4] = {zero, zero, zero, zero};
  for (int mc = 0; mc < MM; mc += 32) {
    v8f t0, t1;
    score_pair(s_mq, mc, bq0, bq1, lo, hi, t0, t1);
    v16h p;
#pragma unroll
    for (int r = 0; r < 8; ++r) {
      p[r]     = (_Float16)(__expf(t0[r] - mx) * inv);
      p[r + 8] = (_Float16)(__expf(t1[r] - mx) * inv);
    }
#pragma unroll
    for (int gg = 0; gg < 4; ++gg) {
      v16h bv = load_b16(s_mvT, MM, gg * 16, mc, lo, hi);
      acc[gg] = WMMA_F16(p, bv, acc[gg]);
    }
  }
#pragma unroll
  for (int gg = 0; gg < 4; ++gg) {
    int d = gg * 16 + lo;
#pragma unroll
    for (int r = 0; r < 8; ++r) {
      int n = nt * 16 + hi * 8 + r;
      featattn[((size_t)b * INNER + (d * 4 + h)) * NN + n] = acc[gg][r];
    }
  }
}

// K6a: map direction, N partitioned (27 x 512). 8 waves/block, each wave one
// 16-col m-tile; the partition's fq slice and fvT slice staged via TDM.
// Produces per-partition local max/sumexp and UNNORMALIZED partial O.
__global__ void k_map_attn_part(const _Float16* __restrict__ fq,
                                const _Float16* __restrict__ mq,
                                const _Float16* __restrict__ fvT,
                                float* __restrict__ pmax,
                                float* __restrict__ psum,
                                float* __restrict__ opart) {
  __shared__ __align__(32) _Float16 s_fq[PS * DH];   // 64KB
  __shared__ __align__(32) _Float16 s_fvT[DH * PS];  // 64KB
  int blk = blockIdx.x;                  // (bh*4 + mgrp)*NP + part
  int part = blk % NP; blk /= NP;
  int mgrp = blk % 4; int bh = blk / 4;
  int wid = threadIdx.x >> 5;
  int mt = mgrp * 8 + wid;
  int lane = threadIdx.x & 31, lo = lane & 15, hi = lane >> 4;
  const _Float16* fq_bh  = fq  + (size_t)bh * NN * DH;
  const _Float16* mq_bh  = mq  + (size_t)bh * MM * DH;
  const _Float16* fvT_bh = fvT + (size_t)bh * DH * NN;

  if (wid == 0) {
    tdm_load_2d(s_fq, fq_bh + (size_t)part * PS * DH, DH, PS, DH); // contiguous slice
    tdm_load_2d(s_fvT, fvT_bh + (size_t)part * PS, PS, DH, NN);    // strided rows
    tdm_wait();
  }
  __syncthreads();

  v16h bm0 = load_b16(mq_bh, DH, mt * 16, 0, lo, hi);
  v16h bm1 = load_b16(mq_bh, DH, mt * 16, 32, lo, hi);

  float mx = -1e30f, sm = 0.f;
  for (int nc = 0; nc < PS; nc += 32) {
    v8f t0, t1;
    score_pair(s_fq, nc, bm0, bm1, lo, hi, t0, t1);
    float cmax = mx;
#pragma unroll
    for (int r = 0; r < 8; ++r) cmax = fmaxf(cmax, fmaxf(t0[r], t1[r]));
    cmax = fmaxf(cmax, __shfl_xor(cmax, 16));
    float cs = 0.f;
#pragma unroll
    for (int r = 0; r < 8; ++r) cs += __expf(t0[r] - cmax) + __expf(t1[r] - cmax);
    cs += __shfl_xor(cs, 16);
    sm = sm * __expf(mx - cmax) + cs;
    mx = cmax;
  }

  v8f zero = {};
  v8f acc[4] = {zero, zero, zero, zero};
  for (int nc = 0; nc < PS; nc += 32) {
    v8f t0, t1;
    score_pair(s_fq, nc, bm0, bm1, lo, hi, t0, t1);
    v16h p;
#pragma unroll
    for (int r = 0; r < 8; ++r) {
      p[r]     = (_Float16)__expf(t0[r] - mx);
      p[r + 8] = (_Float16)__expf(t1[r] - mx);
    }
#pragma unroll
    for (int gg = 0; gg < 4; ++gg) {
      v16h bf = load_b16(s_fvT, PS, gg * 16, nc, lo, hi);
      acc[gg] = WMMA_F16(p, bf, acc[gg]);
    }
  }
  // stats: per-lane m = mt*16+lo (both half-waves hold identical values)
  size_t sidx = ((size_t)bh * NP + part) * MM + mt * 16 + lo;
  pmax[sidx] = mx;
  psum[sidx] = sm;
#pragma unroll
  for (int gg = 0; gg < 4; ++gg) {
    int d = gg * 16 + lo;
#pragma unroll
    for (int r = 0; r < 8; ++r) {
      int m = mt * 16 + hi * 8 + r;
      opart[(((size_t)bh * NP + part) * MM + m) * DH + d] = acc[gg][r];
    }
  }
}

// K6b: combine partition partials into globally-softmaxed map attention out.
__global__ void k_map_combine(const float* __restrict__ pmax,
                              const float* __restrict__ psum,
                              const float* __restrict__ opart,
                              float* __restrict__ mapattn) {
  int t = blockIdx.x * blockDim.x + threadIdx.x;
  if (t >= BATCH * HEADS * DH * MM) return;
  int m = t % MM; int d = (t / MM) % DH; int bh = t / (MM * DH);
  int h = bh % HEADS, b = bh / HEADS;
  float gmax = -1e30f;
  for (int p = 0; p < NP; ++p)
    gmax = fmaxf(gmax, pmax[((size_t)bh * NP + p) * MM + m]);
  float gsum = 0.f, val = 0.f;
  for (int p = 0; p < NP; ++p) {
    size_t si = ((size_t)bh * NP + p) * MM + m;
    float f = __expf(pmax[si] - gmax);
    gsum += psum[si] * f;
    val  += opart[si * DH + d] * f;
  }
  mapattn[((size_t)b * INNER + (d * 4 + h)) * MM + m] = val / gsum;
}

// K8: output pointwise GEMM via WMMA: [128 x 256] @ [256 x NN] -> d_out feat
__global__ void k_out_gemm(const _Float16* __restrict__ pwout_h,
                           const _Float16* __restrict__ dwout,
                           float* __restrict__ outp) {
  int blk = blockIdx.x;
  int nt = blk % (NN / 16); blk /= (NN / 16);
  int ot = blk % 8; int b = blk / 8;
  int lane = threadIdx.x & 31, lo = lane & 15, hi = lane >> 4;
  const _Float16* bsrc = dwout + (size_t)b * NN * INNER;
  v8f acc = {};
#pragma unroll
  for (int kc = 0; kc < INNER; kc += 32) {
    v16h a  = load_a16(pwout_h, INNER, ot * 16, kc, lo, hi);
    v16h bm = load_b16(bsrc, INNER, nt * 16, kc, lo, hi);
    acc = WMMA_F16(a, bm, acc);
  }
  int n = nt * 16 + lo;
#pragma unroll
  for (int r = 0; r < 8; ++r) {
    int o = ot * 16 + hi * 8 + r;
    outp[((size_t)b * OUTC + o) * NN + n] = acc[r];
  }
}

// K9: map output projection (tiny)
__global__ void k_map_out(const float* __restrict__ w,
                          const float* __restrict__ mapattn,
                          float* __restrict__ outm) {
  int t = blockIdx.x * blockDim.x + threadIdx.x;
  if (t >= BATCH * MAPC * MM) return;
  int m = t % MM; int bo = t / MM; int o = bo % MAPC; int b = bo / MAPC;
  float acc = 0.f;
  for (int c = 0; c < INNER; ++c)
    acc += w[(size_t)o * INNER + c] * mapattn[((size_t)b * INNER + c) * MM + m];
  outm[((size_t)b * MAPC + o) * MM + m] = acc;
}

// ---------------------------------------------------------------------------
extern "C" void kernel_launch(void* const* d_in, const int* in_sizes, int n_in,
                              void* d_out, int out_size, void* d_ws, size_t ws_size,
                              hipStream_t stream) {
  (void)in_sizes; (void)n_in; (void)out_size; (void)ws_size;
  const float* feat   = (const float*)d_in[0];
  const float* smap   = (const float*)d_in[1];
  const float* qv_dw  = (const float*)d_in[2];
  const float* qv_pw  = (const float*)d_in[3];
  const float* out_dw = (const float*)d_in[4];
  const float* out_pw = (const float*)d_in[5];
  const float* mqv_w  = (const float*)d_in[6];
  const float* mout_w = (const float*)d_in[7];

  char* ws = (char*)d_ws;
  size_t off = 0;
  auto alloc = [&](size_t bytes) -> void* {
    void* p = ws + off;
    off = (off + bytes + 255) & ~(size_t)255;
    return p;
  };
  _Float16* pwqv_h   = (_Float16*)alloc((size_t)2 * INNER * FEATC * 2);
  _Float16* pwout_h  = (_Float16*)alloc((size_t)OUTC * INNER * 2);
  _Float16* dwfeat   = (_Float16*)alloc((size_t)BATCH * NN * FEATC * 2);
  _Float16* fq       = (_Float16*)alloc((size_t)BATCH * HEADS * NN * DH * 2);
  _Float16* fvT      = (_Float16*)alloc((size_t)BATCH * HEADS * DH * NN * 2);
  _Float16* mqb      = (_Float16*)alloc((size_t)BATCH * HEADS * MM * DH * 2);
  _Float16* mvT      = (_Float16*)alloc((size_t)BATCH * HEADS * DH * MM * 2);
  float*    featattn = (float*)alloc((size_t)BATCH * INNER * NN * 4);
  _Float16* dwout    = (_Float16*)alloc((size_t)BATCH * NN * INNER * 2);
  float*    mapattn  = (float*)alloc((size_t)BATCH * INNER * MM * 4);
  float*    pmax_ws  = (float*)alloc((size_t)BATCH * HEADS * NP * MM * 4);
  float*    psum_ws  = (float*)alloc((size_t)BATCH * HEADS * NP * MM * 4);
  float*    opart_ws = (float*)alloc((size_t)BATCH * HEADS * NP * MM * DH * 4);

  k_cvt_weights<<<(2 * INNER * FEATC + 255) / 256, 256, 0, stream>>>(
      qv_pw, out_pw, pwqv_h, pwout_h);
  k_dw3<<<(BATCH * FEATC * NN + 255) / 256, 256, 0, stream>>>(
      feat, qv_dw, dwfeat, FEATC);
  k_qv_gemm<<<BATCH * 32 * (NN / 16), 32, 0, stream>>>(pwqv_h, dwfeat, fq, fvT);
  k_map_qv<<<(BATCH * 2 * INNER * MM + 255) / 256, 256, 0, stream>>>(
      mqv_w, smap, mqb, mvT);
  k_feat_attn<<<BATCH * HEADS * (NN / 128), 256, 0, stream>>>(
      fq, mqb, mvT, featattn);
  k_map_attn_part<<<BATCH * HEADS * 4 * NP, 256, 0, stream>>>(
      fq, mqb, fvT, pmax_ws, psum_ws, opart_ws);
  k_map_combine<<<(BATCH * HEADS * DH * MM + 255) / 256, 256, 0, stream>>>(
      pmax_ws, psum_ws, opart_ws, mapattn);
  k_dw3<<<(BATCH * INNER * NN + 255) / 256, 256, 0, stream>>>(
      featattn, out_dw, dwout, INNER);
  k_out_gemm<<<BATCH * 8 * (NN / 16), 32, 0, stream>>>(
      pwout_h, dwout, (float*)d_out);
  k_map_out<<<(BATCH * MAPC * MM + 255) / 256, 256, 0, stream>>>(
      mout_w, mapattn, (float*)d_out + (size_t)BATCH * OUTC * NN);
}
